// Linear_Attention_326417514711
// MI455X (gfx1250) — matrix operand order repaired and checked by reading, not compiled, not run on hardware
//
#include <hip/hip_runtime.h>
#include <math.h>

// ---------------------------------------------------------------------------
// Problem constants (from reference)
// ---------------------------------------------------------------------------
#define BB   4
#define LL   4096
#define SS   4096
#define INC  1024
#define NF   1024
#define OUTC 1024
#define NH   16
#define DD   64          // per-head dim
#define MROWS (BB * LL)  // 16384

// ---------------------------------------------------------------------------
// bf16 helpers (raw ushort storage; WMMA fragments via ext vectors)
// ---------------------------------------------------------------------------
typedef __attribute__((ext_vector_type(16))) __bf16 v16bf;
typedef __attribute__((ext_vector_type(8)))  float  v8f;

union FragAB { v16bf v; unsigned u32[8]; unsigned short u16[16]; };
union FragC  { v8f   v; float f[8]; };
union U4x8   { uint4 q; unsigned short u[8]; };

__device__ __forceinline__ unsigned short f32_to_bf16(float x) {
    union { float f; unsigned u; } c; c.f = x;
    unsigned u = c.u;
    u += 0x7FFFu + ((u >> 16) & 1u);   // round-to-nearest-even
    return (unsigned short)(u >> 16);
}
__device__ __forceinline__ unsigned pack2_bf16(float a, float b) {
    return (unsigned)f32_to_bf16(a) | ((unsigned)f32_to_bf16(b) << 16);
}
__device__ __forceinline__ float bf16_to_f32(unsigned short b) {
    union { unsigned u; float f; } c; c.u = ((unsigned)b) << 16;
    return c.f;
}
__device__ __forceinline__ float sigmoidf_(float x) {
    return 1.0f / (1.0f + expf(-x));
}

// ---------------------------------------------------------------------------
// Fragment loaders from LDS tiles stored "fragment-row-major":
//   A tile:  row = M index, contiguous K (stride in ushorts, even)
//   B tile:  row = N index, contiguous K
// Layouts per CDNA5 ISA 7.12.2 (16-bit A 16x32; B 32x16).
// ---------------------------------------------------------------------------
__device__ __forceinline__ void load_frag_a(FragAB& f, const unsigned short* base,
                                            int stride, int lane) {
    const int m  = lane & 15;
    const int kb = (lane >> 4) << 3;          // 0 or 8
    const unsigned short* row = base + m * stride;
#pragma unroll
    for (int j = 0; j < 8; ++j) {
        const int kk = ((j & 4) << 2) + kb + ((j & 3) << 1); // j>=4 -> +16
        f.u32[j] = *(const unsigned*)(row + kk);
    }
}
__device__ __forceinline__ void load_frag_b(FragAB& f, const unsigned short* base,
                                            int stride, int lane) {
    const int n  = lane & 15;
    const int kb = (lane >> 4) << 3;          // 0 or 8
    const unsigned short* row = base + n * stride;
#pragma unroll
    for (int j = 0; j < 8; ++j) {
        const int kk = ((j & 4) << 2) + kb + ((j & 3) << 1); // j>=4 -> +16
        f.u32[j] = *(const unsigned*)(row + kk);
    }
}
__device__ __forceinline__ v8f wmma_bf16(const FragAB& a, const FragAB& b, v8f c) {
    return __builtin_amdgcn_wmma_f32_16x16x32_bf16(
        false, a.v, false, b.v, (short)0, c, false, false);
}

// ---------------------------------------------------------------------------
// Generic GEMM:  out = act(A[MxK] * W[KxN] + bias)  (+ optional mask*f32 out)
// Block tile 128x128, 256 threads (8 wave32), wave tile 64x32.
// ACT: 0 = none, 1 = sigmoid.  A_BF16: A given as bf16 ushort.  MASK_OUT:
// write f32 with per-row mask, else write bf16.
// ---------------------------------------------------------------------------
template <int ACT, bool A_BF16, bool MASK_OUT>
__global__ __launch_bounds__(256)
void gemm_kernel(const float* __restrict__ Af, const unsigned short* __restrict__ Ab,
                 const float* __restrict__ W, const float* __restrict__ bias,
                 const float* __restrict__ mask,
                 unsigned short* __restrict__ outB, float* __restrict__ outF,
                 int M, int N, int K) {
    __shared__ unsigned short lds_a[128][36]; // m-major, K contiguous (72B rows)
    __shared__ unsigned short lds_b[128][36]; // n-major (W tile transposed)

    const int tid  = threadIdx.x;
    const int lane = tid & 31;
    const int wid  = tid >> 5;
    const int wm   = wid >> 2;   // 0..1 -> 64-row strip
    const int wn   = wid & 3;    // 0..3 -> 32-col strip
    const int m0   = blockIdx.y * 128;
    const int n0   = blockIdx.x * 128;

    FragC acc[4][2];
#pragma unroll
    for (int i = 0; i < 4; ++i)
#pragma unroll
        for (int j = 0; j < 2; ++j)
#pragma unroll
            for (int r = 0; r < 8; ++r) acc[i][j].f[r] = 0.0f;

    for (int k0 = 0; k0 < K; k0 += 32) {
        // stage A tile (128x32) -> bf16, m-major; 128-bit loads, 64-bit LDS stores
#pragma unroll
        for (int i = 0; i < 4; ++i) {
            const int idx = tid + i * 256;      // 0..1023
            const int r  = idx >> 3;            // 0..127
            const int c4 = (idx & 7) << 2;      // 0,4,..,28
            const size_t g = (size_t)(m0 + r) * K + (k0 + c4);
            uint2 p;
            if (A_BF16) {
                p = *(const uint2*)(&Ab[g]);
            } else {
                const float4 a4 = *(const float4*)(&Af[g]);
                p.x = pack2_bf16(a4.x, a4.y);
                p.y = pack2_bf16(a4.z, a4.w);
            }
            *(uint2*)(&lds_a[r][c4]) = p;
        }
        // stage W tile (32x128) -> bf16, transposed to n-major; 128-bit loads
#pragma unroll
        for (int i = 0; i < 4; ++i) {
            const int idx = tid + i * 256;      // 0..1023
            const int kk = idx >> 5;            // 0..31
            const int n4 = (idx & 31) << 2;     // 0,4,..,124
            const float4 w4 = *(const float4*)(&W[(size_t)(k0 + kk) * N + (n0 + n4)]);
            lds_b[n4 + 0][kk] = f32_to_bf16(w4.x);
            lds_b[n4 + 1][kk] = f32_to_bf16(w4.y);
            lds_b[n4 + 2][kk] = f32_to_bf16(w4.z);
            lds_b[n4 + 3][kk] = f32_to_bf16(w4.w);
        }
        __syncthreads();

        FragAB afr[4], bfr[2];
#pragma unroll
        for (int ti = 0; ti < 4; ++ti)
            load_frag_a(afr[ti], &lds_a[wm * 64 + ti * 16][0], 36, lane);
#pragma unroll
        for (int tj = 0; tj < 2; ++tj)
            load_frag_b(bfr[tj], &lds_b[wn * 32 + tj * 16][0], 36, lane);
#pragma unroll
        for (int ti = 0; ti < 4; ++ti)
#pragma unroll
            for (int tj = 0; tj < 2; ++tj)
                acc[ti][tj].v = wmma_bf16(afr[ti], bfr[tj], acc[ti][tj].v);
        __syncthreads();
    }

    // epilogue (D layout: col = lane&15, rows = r + 8*(lane>=16))
#pragma unroll
    for (int ti = 0; ti < 4; ++ti) {
#pragma unroll
        for (int tj = 0; tj < 2; ++tj) {
            const int col   = n0 + wn * 32 + tj * 16 + (lane & 15);
            const int rbase = m0 + wm * 64 + ti * 16 + ((lane >> 4) << 3);
            const float bv  = bias[col];
#pragma unroll
            for (int r = 0; r < 8; ++r) {
                const int row = rbase + r;
                float val = acc[ti][tj].f[r] + bv;
                if (ACT == 1) val = sigmoidf_(val);
                if (MASK_OUT)
                    outF[(size_t)row * N + col] = val * mask[row];
                else
                    outB[(size_t)row * N + col] = f32_to_bf16(val);
            }
        }
    }
}

// ---------------------------------------------------------------------------
// ksum[b,h,d] = sum_s Kp[b,s,h*64+d]   (one block per (b,h))
// paired-d b32 loads: each wave covers all 64 d -> 128B coalesced per s
// ---------------------------------------------------------------------------
__global__ __launch_bounds__(256)
void ksum_kernel(const unsigned short* __restrict__ Kp, float* __restrict__ ksum) {
    const int bh = blockIdx.x;
    const int b = bh >> 4, h = bh & 15;
    const int dp = (threadIdx.x & 31) << 1;   // 0,2,..,62
    const int sl = threadIdx.x >> 5;          // 0..7
    float a0 = 0.0f, a1 = 0.0f;
    for (int s = sl; s < SS; s += 8) {
        const unsigned w = *(const unsigned*)(&Kp[(size_t)(b * SS + s) * NF + h * DD + dp]);
        a0 += bf16_to_f32((unsigned short)w);
        a1 += bf16_to_f32((unsigned short)(w >> 16));
    }
    __shared__ float red[8][64];
    red[sl][dp] = a0;
    red[sl][dp + 1] = a1;
    __syncthreads();
    if (threadIdx.x < 64) {
        float s = 0.0f;
#pragma unroll
        for (int i = 0; i < 8; ++i) s += red[i][threadIdx.x];
        ksum[bh * DD + threadIdx.x] = s;
    }
}

// ---------------------------------------------------------------------------
// KV[b,h,d,e] += sum_{s in chunk} Kp[b,s,hd] * Vp[b,s,he]
// grid (B*H, S/256), 128 threads = 4 waves; wave w -> d rows [16w,16w+16)
// ---------------------------------------------------------------------------
__global__ __launch_bounds__(128)
void kv_kernel(const unsigned short* __restrict__ Kp,
               const unsigned short* __restrict__ Vp,
               float* __restrict__ KVbuf) {
    __shared__ unsigned short lds_k[64][36]; // d-major [d][s]
    __shared__ unsigned short lds_v[64][36]; // e-major [e][s]
    const int bh = blockIdx.x;
    const int b = bh >> 4, h = bh & 15;
    const int s0base = blockIdx.y * 256;
    const int tid = threadIdx.x;
    const int lane = tid & 31;
    const int wid = tid >> 5;

    FragC acc[4];
#pragma unroll
    for (int j = 0; j < 4; ++j)
#pragma unroll
        for (int r = 0; r < 8; ++r) acc[j].f[r] = 0.0f;

    for (int ss = 0; ss < 256; ss += 32) {
        // 128-bit loads over contiguous d, transposed u16 stores into LDS
#pragma unroll
        for (int i = 0; i < 2; ++i) {
            const int idx = tid + i * 128;   // 0..255
            const int s  = idx >> 3;          // 0..31
            const int d8 = (idx & 7) << 3;    // 0,8,..,56
            const size_t g = (size_t)(b * SS + s0base + ss + s) * NF + h * DD + d8;
            U4x8 kq, vq;
            kq.q = *(const uint4*)(&Kp[g]);
            vq.q = *(const uint4*)(&Vp[g]);
#pragma unroll
            for (int j = 0; j < 8; ++j) {
                lds_k[d8 + j][s] = kq.u[j];
                lds_v[d8 + j][s] = vq.u[j];
            }
        }
        __syncthreads();
        FragAB afr, bfr[4];
        load_frag_a(afr, &lds_k[wid * 16][0], 36, lane);
#pragma unroll
        for (int tj = 0; tj < 4; ++tj)
            load_frag_b(bfr[tj], &lds_v[tj * 16][0], 36, lane);
#pragma unroll
        for (int tj = 0; tj < 4; ++tj)
            acc[tj].v = wmma_bf16(afr, bfr[tj], acc[tj].v);
        __syncthreads();
    }
#pragma unroll
    for (int tj = 0; tj < 4; ++tj) {
        const int e = tj * 16 + (lane & 15);
        const int dbase = wid * 16 + ((lane >> 4) << 3);
#pragma unroll
        for (int r = 0; r < 8; ++r)
            atomicAdd(&KVbuf[((size_t)bh * DD + (dbase + r)) * DD + e], acc[tj].f[r]);
    }
}

// ---------------------------------------------------------------------------
// Z[b,h,l] = 1 / sum_d q[l,d] * (ksum[d] + 1e-6)
// ---------------------------------------------------------------------------
__global__ __launch_bounds__(256)
void z_kernel(const unsigned short* __restrict__ Qp, const float* __restrict__ ksum,
              float* __restrict__ Z) {
    const int idx = blockIdx.x * 256 + threadIdx.x;   // B*H*L
    if (idx >= BB * NH * LL) return;
    const int bh = idx >> 12, l = idx & 4095;
    const int b = bh >> 4, h = bh & 15;
    const unsigned short* q = &Qp[(size_t)(b * LL + l) * NF + h * DD];
    const float* ks = &ksum[bh * DD];
    float acc = 0.0f;
#pragma unroll
    for (int i = 0; i < 8; ++i) {
        U4x8 c;
        c.q = *(const uint4*)(&q[i * 8]);
#pragma unroll
        for (int j = 0; j < 8; ++j)
            acc += bf16_to_f32(c.u[j]) * (ks[i * 8 + j] + 1e-6f);
    }
    Z[idx] = 1.0f / acc;
}

// ---------------------------------------------------------------------------
// Xb[b,l,h*64+e] = Z[b,h,l] * sum_d q[l,d] * KV[d,e]
// grid (B*H, L/256), 256 threads = 8 waves; wave -> 32 rows x 64 cols
// ---------------------------------------------------------------------------
__global__ __launch_bounds__(256)
void x_kernel(const unsigned short* __restrict__ Qp, const float* __restrict__ KVbuf,
              const float* __restrict__ Z, unsigned short* __restrict__ Xb) {
    __shared__ unsigned short lds_q[256][72];  // l-major [l][d], 144B rows (16B aligned)
    __shared__ unsigned short lds_kv[64][72];  // e-major [e][d]
    const int bh = blockIdx.x;
    const int b = bh >> 4, h = bh & 15;
    const int l0 = blockIdx.y * 256;
    const int tid = threadIdx.x;
    const int lane = tid & 31;
    const int wid = tid >> 5;

    // stage KV (f32 -> bf16, transpose to e-major); 128-bit loads
#pragma unroll
    for (int i = 0; i < 4; ++i) {
        const int idx = tid + i * 256;   // 0..1023
        const int d  = idx >> 4;          // 0..63
        const int e4 = (idx & 15) << 2;   // 0,4,..,60
        const float4 kv4 = *(const float4*)(&KVbuf[((size_t)bh * DD + d) * DD + e4]);
        lds_kv[e4 + 0][d] = f32_to_bf16(kv4.x);
        lds_kv[e4 + 1][d] = f32_to_bf16(kv4.y);
        lds_kv[e4 + 2][d] = f32_to_bf16(kv4.z);
        lds_kv[e4 + 3][d] = f32_to_bf16(kv4.w);
    }
    // stage q rows; 128-bit global loads + 128-bit LDS stores
#pragma unroll
    for (int i = 0; i < 8; ++i) {
        const int idx = tid + i * 256;   // 0..2047
        const int r  = idx >> 3;          // 0..255
        const int c8 = (idx & 7) << 3;    // 0,8,..,56
        const uint4 qv = *(const uint4*)(&Qp[(size_t)(b * LL + l0 + r) * NF + h * DD + c8]);
        *(uint4*)(&lds_q[r][c8]) = qv;
    }
    __syncthreads();

    FragC acc[2][4];
#pragma unroll
    for (int i = 0; i < 2; ++i)
#pragma unroll
        for (int j = 0; j < 4; ++j)
#pragma unroll
            for (int r = 0; r < 8; ++r) acc[i][j].f[r] = 0.0f;

#pragma unroll
    for (int k0 = 0; k0 < DD; k0 += 32) {
        FragAB afr[2], bfr[4];
#pragma unroll
        for (int ti = 0; ti < 2; ++ti)
            load_frag_a(afr[ti], &lds_q[wid * 32 + ti * 16][k0], 72, lane);
#pragma unroll
        for (int tj = 0; tj < 4; ++tj)
            load_frag_b(bfr[tj], &lds_kv[tj * 16][k0], 72, lane);
#pragma unroll
        for (int ti = 0; ti < 2; ++ti)
#pragma unroll
            for (int tj = 0; tj < 4; ++tj)
                acc[ti][tj].v = wmma_bf16(afr[ti], bfr[tj], acc[ti][tj].v);
    }

#pragma unroll
    for (int ti = 0; ti < 2; ++ti) {
#pragma unroll
        for (int tj = 0; tj < 4; ++tj) {
            const int e = tj * 16 + (lane & 15);
            const int rbase = wid * 32 + ti * 16 + ((lane >> 4) << 3);
#pragma unroll
            for (int r = 0; r < 8; ++r) {
                const int lrow = l0 + rbase + r;
                const float val = acc[ti][tj].f[r] * Z[(size_t)bh * LL + lrow];
                Xb[(size_t)(b * LL + lrow) * NF + h * DD + e] = f32_to_bf16(val);
            }
        }
    }
}

// ---------------------------------------------------------------------------
// Launch
// ---------------------------------------------------------------------------
extern "C" void kernel_launch(void* const* d_in, const int* in_sizes, int n_in,
                              void* d_out, int out_size, void* d_ws, size_t ws_size,
                              hipStream_t stream) {
    (void)in_sizes; (void)n_in; (void)out_size; (void)ws_size;
    const float* Q    = (const float*)d_in[0];
    const float* Kin  = (const float*)d_in[1];
    const float* Vin  = (const float*)d_in[2];
    const float* mask = (const float*)d_in[3];
    const float* Wq = (const float*)d_in[4];  const float* bq = (const float*)d_in[5];
    const float* Wk = (const float*)d_in[6];  const float* bk = (const float*)d_in[7];
    const float* Wv = (const float*)d_in[8];  const float* bv = (const float*)d_in[9];
    const float* Wo = (const float*)d_in[10]; const float* bo = (const float*)d_in[11];
    float* out = (float*)d_out;

    char* ws = (char*)d_ws;
    size_t off = 0;
    auto take = [&](size_t bytes) { char* p = ws + off; off += (bytes + 255) & ~(size_t)255; return p; };
    unsigned short* Qp = (unsigned short*)take((size_t)MROWS * NF * 2);
    unsigned short* Kp = (unsigned short*)take((size_t)MROWS * NF * 2);
    unsigned short* Vp = (unsigned short*)take((size_t)MROWS * NF * 2);
    unsigned short* Xb = (unsigned short*)take((size_t)MROWS * NF * 2);
    float* KVbuf = (float*)take((size_t)BB * NH * DD * DD * 4);
    float* ksum  = (float*)take((size_t)BB * NH * DD * 4);
    float* Zbuf  = (float*)take((size_t)BB * NH * LL * 4);

    const dim3 gGemm(NF / 128, MROWS / 128);
    // q = sigmoid(Q*Wq+bq), k = sigmoid(K*Wk+bk), v = V*Wv+bv   (bf16 out)
    gemm_kernel<1, false, false><<<gGemm, 256, 0, stream>>>(Q, nullptr, Wq, bq, nullptr, Qp, nullptr, MROWS, NF, INC);
    gemm_kernel<1, false, false><<<gGemm, 256, 0, stream>>>(Kin, nullptr, Wk, bk, nullptr, Kp, nullptr, MROWS, NF, INC);
    gemm_kernel<0, false, false><<<gGemm, 256, 0, stream>>>(Vin, nullptr, Wv, bv, nullptr, Vp, nullptr, MROWS, NF, INC);

    // ksum over S
    ksum_kernel<<<BB * NH, 256, 0, stream>>>(Kp, ksum);

    // KV = k^T v (accumulated via f32 atomics; buffer zeroed every launch)
    hipMemsetAsync(KVbuf, 0, (size_t)BB * NH * DD * DD * 4, stream);
    kv_kernel<<<dim3(BB * NH, SS / 256), 128, 0, stream>>>(Kp, Vp, KVbuf);

    // Z normalizer
    z_kernel<<<(BB * NH * LL) / 256, 256, 0, stream>>>(Qp, ksum, Zbuf);

    // x = Z * (q @ KV), packed back to [B, L, H*D] bf16
    x_kernel<<<dim3(BB * NH, LL / 256), 256, 0, stream>>>(Qp, KVbuf, Zbuf, Xb);

    // out = (x @ Wo + bo) * mask  (f32)
    gemm_kernel<0, true, true><<<gGemm, 256, 0, stream>>>(nullptr, Xb, Wo, bo, mask, nullptr, out, MROWS, OUTC, NF);
}
